// BiLSTMCRF_47493748359581
// MI455X (gfx1250) — matrix operand order repaired and checked by reading, not compiled, not run on hardware
//
#include <hip/hip_runtime.h>
#include <stdint.h>

// ---------------------------------------------------------------------------
// BiLSTM-CRF for MI455X (gfx1250, wave32).
//   * bf16 WMMA (v_wmma_f32_16x16x32_bf16) for all GEMMs + recurrent matmuls
//   * async global->LDS staging (double buffered) in the big GEMMs
//   * persistent one-WGP LSTM scan: h in LDS, c in accumulator VGPRs,
//     i/f/g/o gate columns of one hidden slice owned by the same wave
// ---------------------------------------------------------------------------

#define USE_ASYNC_LDS 1

typedef __bf16 bf16_t;
typedef __attribute__((ext_vector_type(16))) __bf16 v16bf;
typedef __attribute__((ext_vector_type(8)))  float  v8f;

#define BQ   32          // batch
#define TQ   256         // seq len
#define EQ   300         // embed dim
#define EP   320         // embed dim padded to K%32==0
#define HDQ  256         // hidden per direction
#define HQ   512         // 2*HDQ
#define GQ   1024        // 4*HDQ gates
#define LQ   9           // labels
#define MR   (BQ*TQ)     // 8192 rows, time-major (r = t*B + b)

#if USE_ASYNC_LDS
#define ASYNC_LDS_B128(ldsOff, gaddr) \
  asm volatile("global_load_async_to_lds_b128 %0, %1, off" :: "v"(ldsOff), "v"(gaddr) : "memory")
#define WAIT_ASYNC(n) asm volatile("s_wait_asynccnt %0" :: "i"(n) : "memory")
#endif

union FragU { uint4 u[2]; v16bf v; };

// A-matrix 16x32 bf16 fragment (ISA 7.12.2): lanes 0-15 hold M=lane, K=0..7 then
// 16..23; lanes 16-31 hold K=8..15 then 24..31  -> two b128 loads per lane.
__device__ __forceinline__ v16bf load_frag_a(const bf16_t* t, int rowBase, int lane,
                                             int stride, int kOff) {
  int row  = rowBase + (lane & 15);
  int half = lane >> 4;
  FragU f;
  const bf16_t* p = t + (size_t)row * stride + kOff + half * 8;
  f.u[0] = *(const uint4*)p;
  f.u[1] = *(const uint4*)(p + 16);
  return f.v;
}

// B-matrix 32x16 bf16 fragment: lane n holds column n; lanes 0-15 K=0..7 then
// 16..23; lanes 16-31 K=8..15 then 24..31 -> two b128 loads per lane.
// Row-major (N,K) weights feed this directly (computes A * W^T).
__device__ __forceinline__ v16bf load_frag_b(const bf16_t* t, int rowBase, int lane,
                                             int stride, int kOff) {
  int row  = rowBase + (lane & 15);
  int half = lane >> 4;
  FragU f;
  const bf16_t* p = t + (size_t)row * stride + kOff + half * 8;
  f.u[0] = *(const uint4*)p;
  f.u[1] = *(const uint4*)(p + 16);
  return f.v;
}

__device__ __forceinline__ v8f wmma_bf16(v16bf a, v16bf b, v8f c) {
  return __builtin_amdgcn_wmma_f32_16x16x32_bf16(false, a, false, b, (short)0, c,
                                                 false, false);
}

__device__ __forceinline__ float sigf(float x) { return 1.0f / (1.0f + __expf(-x)); }

// ---------------------------------------------------------------------------
// Weight prep: f32 -> bf16 with zero padding (rows to Nd, cols to Kd)
// ---------------------------------------------------------------------------
__global__ void pad_convert_kernel(const float* __restrict__ src, bf16_t* __restrict__ dst,
                                   int Ns, int Ks, int Nd, int Kd) {
  size_t i = (size_t)blockIdx.x * blockDim.x + threadIdx.x;
  if (i >= (size_t)Nd * Kd) return;
  int k = (int)(i % Kd);
  int n = (int)(i / Kd);
  float v = (n < Ns && k < Ks) ? src[(size_t)n * Ks + k] : 0.0f;
  dst[i] = (bf16_t)v;
}

__global__ void bias_sum_kernel(const float* __restrict__ a, const float* __restrict__ b,
                                float* __restrict__ o, int n) {
  int i = blockIdx.x * blockDim.x + threadIdx.x;
  if (i < n) o[i] = a[i] + b[i];
}

__global__ void pad_bias_kernel(const float* __restrict__ a, float* __restrict__ o,
                                int ns, int nd) {
  int i = blockIdx.x * blockDim.x + threadIdx.x;
  if (i < nd) o[i] = (i < ns) ? a[i] : 0.0f;
}

// ---------------------------------------------------------------------------
// Embedding gather -> bf16, time-major rows (t*B + b), K padded 300->320
// ---------------------------------------------------------------------------
__global__ void embed_kernel(const int* __restrict__ ids, const float* __restrict__ emb,
                             bf16_t* __restrict__ x0) {
  size_t i = (size_t)blockIdx.x * blockDim.x + threadIdx.x;
  if (i >= (size_t)MR * EP) return;
  int    e = (int)(i % EP);
  size_t r = i / EP;
  int t = (int)(r / BQ), b = (int)(r % BQ);
  float val = 0.0f;
  if (e < EQ) {
    int id = ids[b * TQ + t];
    val = emb[(size_t)id * EQ + e];
  }
  x0[i] = (bf16_t)val;
}

// ---------------------------------------------------------------------------
// GEMM: C[M x Nout](f32) = A[M x K](bf16, lda) * Bw[N x K]^T(bf16, ldb) + bias
// Block 256 threads = 8 waves in a 2(M) x 4(N) grid; WG tile 64M x 128N;
// each wave computes a 32x32 tile as 2x2 WMMA fragments. K staged through
// double-buffered LDS with async global->LDS b128 copies.
// ---------------------------------------------------------------------------
__global__ void __launch_bounds__(256)
gemm_bf16_kernel(const bf16_t* __restrict__ A, const bf16_t* __restrict__ Bw,
                 const float* __restrict__ bias, float* __restrict__ C,
                 int K, int lda, int ldb, int ldc, int Nout) {
  __shared__ __align__(16) bf16_t As[2][64 * 32];
  __shared__ __align__(16) bf16_t Bs[2][128 * 32];

  const int tid  = threadIdx.x;
  const int lane = tid & 31;
  const int wid  = tid >> 5;
  const int mw   = (wid & 1) * 32;
  const int nw   = (wid >> 1) * 32;
  const int m0   = blockIdx.x * 64;
  const int n0   = blockIdx.y * 128;
  const int NK   = K >> 5;

  v8f acc00 = {}, acc01 = {}, acc10 = {}, acc11 = {};

  const int arow = tid >> 2, acb = tid & 3;   // A tile: 64 rows x 4 16B chunks

  auto issue = [&](int kc, int buf) {
    const bf16_t* ga = A + (size_t)(m0 + arow) * lda + kc * 32 + acb * 8;
#if USE_ASYNC_LDS
    ASYNC_LDS_B128((uint32_t)(uintptr_t)&As[buf][arow * 32 + acb * 8],
                   (uint64_t)(uintptr_t)ga);
#else
    *(uint4*)&As[buf][arow * 32 + acb * 8] = *(const uint4*)ga;
#endif
    __builtin_prefetch(ga + 32, 0, 0);        // speculative prefetch of kc+1
#pragma unroll
    for (int j = 0; j < 2; ++j) {             // B tile: 128 rows x 4 chunks
      int c2 = tid + j * 256;
      int r2 = c2 >> 2, cb2 = c2 & 3;
      const bf16_t* gb = Bw + (size_t)(n0 + r2) * ldb + kc * 32 + cb2 * 8;
#if USE_ASYNC_LDS
      ASYNC_LDS_B128((uint32_t)(uintptr_t)&Bs[buf][r2 * 32 + cb2 * 8],
                     (uint64_t)(uintptr_t)gb);
#else
      *(uint4*)&Bs[buf][r2 * 32 + cb2 * 8] = *(const uint4*)gb;
#endif
    }
  };

  issue(0, 0);
  for (int kc = 0; kc < NK; ++kc) {
    int buf = kc & 1;
    if (kc + 1 < NK) {
      issue(kc + 1, buf ^ 1);   // 3 async b128 per thread, in flight over compute
#if USE_ASYNC_LDS
      WAIT_ASYNC(3);            // own 3 loads of tile kc are done (in-order)
#endif
    } else {
#if USE_ASYNC_LDS
      WAIT_ASYNC(0);
#endif
    }
    __syncthreads();            // all waves' tile-kc data visible in LDS

    v16bf a0 = load_frag_a(&As[buf][0], mw,      lane, 32, 0);
    v16bf a1 = load_frag_a(&As[buf][0], mw + 16, lane, 32, 0);
    v16bf b0 = load_frag_b(&Bs[buf][0], nw,      lane, 32, 0);
    v16bf b1 = load_frag_b(&Bs[buf][0], nw + 16, lane, 32, 0);
    acc00 = wmma_bf16(a0, b0, acc00);
    acc01 = wmma_bf16(a0, b1, acc01);
    acc10 = wmma_bf16(a1, b0, acc10);
    acc11 = wmma_bf16(a1, b1, acc11);

    __syncthreads();            // reads done before next issue overwrites buf
  }

  // C/D layout: lanes 0-15 -> M = v, lanes 16-31 -> M = 8+v; N = lane&15
#pragma unroll
  for (int mi = 0; mi < 2; ++mi) {
#pragma unroll
    for (int ni = 0; ni < 2; ++ni) {
      v8f a = (mi == 0) ? (ni == 0 ? acc00 : acc01) : (ni == 0 ? acc10 : acc11);
      int col   = n0 + nw + ni * 16 + (lane & 15);
      int rbase = m0 + mw + mi * 16 + ((lane >> 4) << 3);
      if (col < Nout) {
        float bb = bias ? bias[col] : 0.0f;
#pragma unroll
        for (int v = 0; v < 8; ++v)
          C[(size_t)(rbase + v) * ldc + col] = a[v] + bb;
      }
    }
  }
}

// ---------------------------------------------------------------------------
// Persistent LSTM scan, one direction. Single WG of 8 waves.
// Gpre: (T*B, 1024) f32 precomputed x@Wih^T + (bih+bhh), time-major.
// Whh:  (1024, 256) bf16 row-major (L2-resident, streamed every step).
// Wave w owns hidden slice [w*32, w*32+32): its 16 accum fragments are the
// i/f/g/o gate columns {j, 256+j, 512+j, 768+j} for that slice, so the cell
// update is register-only. h broadcast via LDS (32x256 bf16) each step.
// ---------------------------------------------------------------------------
__global__ void __launch_bounds__(256)
lstm_scan_kernel(const float* __restrict__ Gpre, const bf16_t* __restrict__ Whh,
                 bf16_t* __restrict__ Hout, int hstride, int colOff, int reverse) {
  __shared__ __align__(16) bf16_t hbuf[BQ * HDQ];   // 16 KB

  const int tid  = threadIdx.x;
  const int lane = tid & 31;
  const int wid  = tid >> 5;
  const int hs   = wid * 32;     // hidden slice base for this wave

  for (int i = tid; i < BQ * HDQ; i += 256) hbuf[i] = (bf16_t)0.0f;
  __syncthreads();

  v8f cst[2][2] = {};            // cell state c, same frag layout as gates

  for (int step = 0; step < TQ; ++step) {
    const int tt = reverse ? (TQ - 1 - step) : step;

    v8f acc[2][2][4] = {};       // [mi][ni][gate]
    for (int kc = 0; kc < 8; ++kc) {           // K = 256 = 8 x 32
      v16bf a0 = load_frag_a(hbuf, 0,  lane, HDQ, kc * 32);
      v16bf a1 = load_frag_a(hbuf, 16, lane, HDQ, kc * 32);
#pragma unroll
      for (int ni = 0; ni < 2; ++ni) {
#pragma unroll
        for (int g = 0; g < 4; ++g) {
          v16bf bw = load_frag_b(Whh, g * HDQ + hs + ni * 16, lane, HDQ, kc * 32);
          acc[0][ni][g] = wmma_bf16(a0, bw, acc[0][ni][g]);
          acc[1][ni][g] = wmma_bf16(a1, bw, acc[1][ni][g]);
        }
      }
    }

    bf16_t hreg[2][2][8];
#pragma unroll
    for (int mi = 0; mi < 2; ++mi) {
#pragma unroll
      for (int ni = 0; ni < 2; ++ni) {
        int jc = hs + ni * 16 + (lane & 15);
#pragma unroll
        for (int v = 0; v < 8; ++v) {
          int    brow = mi * 16 + ((lane >> 4) << 3) + v;
          size_t rr   = ((size_t)tt * BQ + brow) * GQ;
          float gi = Gpre[rr + 0 * HDQ + jc] + acc[mi][ni][0][v];
          float gf = Gpre[rr + 1 * HDQ + jc] + acc[mi][ni][1][v];
          float gg = Gpre[rr + 2 * HDQ + jc] + acc[mi][ni][2][v];
          float go = Gpre[rr + 3 * HDQ + jc] + acc[mi][ni][3][v];
          float c  = sigf(gf) * cst[mi][ni][v] + sigf(gi) * tanhf(gg);
          float h  = sigf(go) * tanhf(c);
          cst[mi][ni][v]  = c;
          hreg[mi][ni][v] = (bf16_t)h;
          Hout[((size_t)tt * BQ + brow) * hstride + colOff + jc] = hreg[mi][ni][v];
        }
      }
    }

    __syncthreads();   // all waves finished reading hbuf for this step
#pragma unroll
    for (int mi = 0; mi < 2; ++mi) {
#pragma unroll
      for (int ni = 0; ni < 2; ++ni) {
        int jc = hs + ni * 16 + (lane & 15);
#pragma unroll
        for (int v = 0; v < 8; ++v) {
          int brow = mi * 16 + ((lane >> 4) << 3) + v;
          hbuf[brow * HDQ + jc] = hreg[mi][ni][v];
        }
      }
    }
    __syncthreads();   // new h visible before next step's matmul
  }
}

// ---------------------------------------------------------------------------
// em scatter: ws em (time-major, ldc=16) -> d_out em region (B,T,9)
// ---------------------------------------------------------------------------
__global__ void emcopy_kernel(const float* __restrict__ em16, float* __restrict__ demOut) {
  int i = blockIdx.x * blockDim.x + threadIdx.x;
  if (i >= MR * LQ) return;
  int l = i % LQ;
  int r = i / LQ;
  int b = r % BQ, t = r / BQ;
  demOut[((size_t)b * TQ + t) * LQ + l] = em16[(size_t)r * 16 + l];
}

// ---------------------------------------------------------------------------
// CRF forward (logZ): one block, thread (b, l), alpha in LDS, scan over T
// ---------------------------------------------------------------------------
__global__ void __launch_bounds__(320)
crf_logz_kernel(const float* __restrict__ em16, const int* __restrict__ mask,
                const float* __restrict__ start, const float* __restrict__ trans,
                const float* __restrict__ endv, float* __restrict__ logZ) {
  __shared__ float alpha[LQ][BQ];
  __shared__ float trs[LQ][LQ];
  int b = threadIdx.x, l = threadIdx.y;
  if (b < LQ) trs[b][l] = trans[b * LQ + l];
  alpha[l][b] = start[l] + em16[(size_t)b * 16 + l];
  __syncthreads();
  for (int t = 1; t < TQ; ++t) {
    float mx = -1e30f;
#pragma unroll
    for (int k = 0; k < LQ; ++k) mx = fmaxf(mx, alpha[k][b] + trs[k][l]);
    float s = 0.0f;
#pragma unroll
    for (int k = 0; k < LQ; ++k) s += __expf(alpha[k][b] + trs[k][l] - mx);
    float nxt = mx + __logf(s) + em16[((size_t)t * BQ + b) * 16 + l];
    float neu = mask[b * TQ + t] ? nxt : alpha[l][b];
    __syncthreads();
    alpha[l][b] = neu;
    __syncthreads();
  }
  if (l == 0) {
    float mx = -1e30f;
    for (int k = 0; k < LQ; ++k) mx = fmaxf(mx, alpha[k][b] + endv[k]);
    float s = 0.0f;
    for (int k = 0; k < LQ; ++k) s += __expf(alpha[k][b] + endv[k] - mx);
    logZ[b] = mx + __logf(s);
  }
}

// ---------------------------------------------------------------------------
// CRF gold score + loss = -sum(score - logZ)
// ---------------------------------------------------------------------------
__global__ void crf_score_loss_kernel(const float* __restrict__ em16,
                                      const int* __restrict__ labels,
                                      const int* __restrict__ mask,
                                      const float* __restrict__ start,
                                      const float* __restrict__ trans,
                                      const float* __restrict__ endv,
                                      const float* __restrict__ logZ,
                                      float* __restrict__ lossOut) {
  __shared__ float red[BQ];
  int b    = threadIdx.x;
  int prev = labels[b * TQ + 0];
  float s  = start[prev] + em16[(size_t)b * 16 + prev];
  int msum = mask[b * TQ + 0];
  for (int t = 1; t < TQ; ++t) {
    int   tg = labels[b * TQ + t];
    int   m  = mask[b * TQ + t];
    float e  = em16[((size_t)t * BQ + b) * 16 + tg];
    s   += (trans[prev * LQ + tg] + e) * (float)m;
    msum += m;
    prev = tg;
  }
  int last = labels[b * TQ + (msum - 1)];
  s += endv[last];
  red[b] = s - logZ[b];
  __syncthreads();
  if (b == 0) {
    float acc = 0.0f;
    for (int i = 0; i < BQ; ++i) acc += red[i];
    lossOut[0] = -acc;
  }
}

// ---------------------------------------------------------------------------
// Viterbi forward (max + backpointers) and backtrack
// ---------------------------------------------------------------------------
__global__ void __launch_bounds__(320)
viterbi_fwd_kernel(const float* __restrict__ em16, const int* __restrict__ mask,
                   const float* __restrict__ start, const float* __restrict__ trans,
                   const float* __restrict__ endv, int* __restrict__ bp,
                   int* __restrict__ lastTag) {
  __shared__ float alpha[LQ][BQ];
  __shared__ float trs[LQ][LQ];
  int b = threadIdx.x, l = threadIdx.y;
  if (b < LQ) trs[b][l] = trans[b * LQ + l];
  alpha[l][b] = start[l] + em16[(size_t)b * 16 + l];
  __syncthreads();
  for (int t = 1; t < TQ; ++t) {
    float best = -1e30f;
    int   arg  = 0;
#pragma unroll
    for (int k = 0; k < LQ; ++k) {
      float v = alpha[k][b] + trs[k][l];
      if (v > best) { best = v; arg = k; }
    }
    bp[((size_t)t * BQ + b) * 16 + l] = arg;
    float nxt = best + em16[((size_t)t * BQ + b) * 16 + l];
    float neu = mask[b * TQ + t] ? nxt : alpha[l][b];
    __syncthreads();
    alpha[l][b] = neu;
    __syncthreads();
  }
  if (l == 0) {
    float best = -1e30f;
    int   arg  = 0;
    for (int k = 0; k < LQ; ++k) {
      float v = alpha[k][b] + endv[k];
      if (v > best) { best = v; arg = k; }
    }
    lastTag[b] = arg;
  }
}

__global__ void viterbi_back_kernel(const int* __restrict__ bp,
                                    const int* __restrict__ lastTag,
                                    const int* __restrict__ mask,
                                    float* __restrict__ preds) {
  int b   = threadIdx.x;
  int tag = lastTag[b];
  for (int t = TQ - 1; t >= 1; --t) {
    preds[b * TQ + t] = (float)tag;
    int pv = bp[((size_t)t * BQ + b) * 16 + tag];
    tag = mask[b * TQ + t] ? pv : tag;
  }
  preds[b * TQ + 0] = (float)tag;
}

// ---------------------------------------------------------------------------
// Host launcher
// ---------------------------------------------------------------------------
extern "C" void kernel_launch(void* const* d_in, const int* in_sizes, int n_in,
                              void* d_out, int out_size, void* d_ws, size_t ws_size,
                              hipStream_t stream) {
  (void)in_sizes; (void)n_in; (void)out_size; (void)ws_size;

  const int*   ids    = (const int*)  d_in[0];
  const int*   amask  = (const int*)  d_in[1];
  const int*   labels = (const int*)  d_in[2];
  const float* emb    = (const float*)d_in[3];
  const float* w_ih0f = (const float*)d_in[4];
  const float* w_hh0f = (const float*)d_in[5];
  const float* b_ih0f = (const float*)d_in[6];
  const float* b_hh0f = (const float*)d_in[7];
  const float* w_ih0b = (const float*)d_in[8];
  const float* w_hh0b = (const float*)d_in[9];
  const float* b_ih0b = (const float*)d_in[10];
  const float* b_hh0b = (const float*)d_in[11];
  const float* w_ih1f = (const float*)d_in[12];
  const float* w_hh1f = (const float*)d_in[13];
  const float* b_ih1f = (const float*)d_in[14];
  const float* b_hh1f = (const float*)d_in[15];
  const float* w_ih1b = (const float*)d_in[16];
  const float* w_hh1b = (const float*)d_in[17];
  const float* b_ih1b = (const float*)d_in[18];
  const float* b_hh1b = (const float*)d_in[19];
  const float* w_out  = (const float*)d_in[20];
  const float* b_out  = (const float*)d_in[21];
  const float* c_start= (const float*)d_in[22];
  const float* c_end  = (const float*)d_in[23];
  const float* c_trans= (const float*)d_in[24];

  float* out = (float*)d_out;              // [loss(1) | em(B*T*9) | preds(B*T)]

  char*  ws  = (char*)d_ws;
  size_t off = 0;
  auto alloc = [&](size_t bytes) -> void* {
    void* p = ws + off;
    off = (off + bytes + 255) & ~(size_t)255;
    return p;
  };

  bf16_t* WIH0F = (bf16_t*)alloc((size_t)GQ * EP * 2);
  bf16_t* WIH0B = (bf16_t*)alloc((size_t)GQ * EP * 2);
  bf16_t* WIH1F = (bf16_t*)alloc((size_t)GQ * HQ * 2);
  bf16_t* WIH1B = (bf16_t*)alloc((size_t)GQ * HQ * 2);
  bf16_t* WHH0F = (bf16_t*)alloc((size_t)GQ * HDQ * 2);
  bf16_t* WHH0B = (bf16_t*)alloc((size_t)GQ * HDQ * 2);
  bf16_t* WHH1F = (bf16_t*)alloc((size_t)GQ * HDQ * 2);
  bf16_t* WHH1B = (bf16_t*)alloc((size_t)GQ * HDQ * 2);
  bf16_t* WOUT  = (bf16_t*)alloc((size_t)128 * HQ * 2);
  float*  BIAS0F= (float*) alloc(GQ * 4);
  float*  BIAS0B= (float*) alloc(GQ * 4);
  float*  BIAS1F= (float*) alloc(GQ * 4);
  float*  BIAS1B= (float*) alloc(GQ * 4);
  float*  BOUT16= (float*) alloc(16 * 4);
  bf16_t* X0    = (bf16_t*)alloc((size_t)MR * EP * 2);
  bf16_t* H0    = (bf16_t*)alloc((size_t)MR * HQ * 2);
  bf16_t* H1    = (bf16_t*)alloc((size_t)MR * HQ * 2);
  float*  GF    = (float*) alloc((size_t)MR * GQ * 4);
  float*  GB    = (float*) alloc((size_t)MR * GQ * 4);
  float*  EM    = (float*) alloc((size_t)MR * 16 * 4);
  float*  LOGZ  = (float*) alloc(BQ * 4);
  int*    BP    = (int*)   alloc((size_t)TQ * BQ * 16 * 4);
  int*    LAST  = (int*)   alloc(BQ * 4);

  auto cdiv = [](size_t a, size_t b) { return (unsigned)((a + b - 1) / b); };

  // --- weight prep (bf16, padded) ---
  pad_convert_kernel<<<cdiv((size_t)GQ*EP,256),256,0,stream>>>(w_ih0f, WIH0F, GQ, EQ, GQ, EP);
  pad_convert_kernel<<<cdiv((size_t)GQ*EP,256),256,0,stream>>>(w_ih0b, WIH0B, GQ, EQ, GQ, EP);
  pad_convert_kernel<<<cdiv((size_t)GQ*HQ,256),256,0,stream>>>(w_ih1f, WIH1F, GQ, HQ, GQ, HQ);
  pad_convert_kernel<<<cdiv((size_t)GQ*HQ,256),256,0,stream>>>(w_ih1b, WIH1B, GQ, HQ, GQ, HQ);
  pad_convert_kernel<<<cdiv((size_t)GQ*HDQ,256),256,0,stream>>>(w_hh0f, WHH0F, GQ, HDQ, GQ, HDQ);
  pad_convert_kernel<<<cdiv((size_t)GQ*HDQ,256),256,0,stream>>>(w_hh0b, WHH0B, GQ, HDQ, GQ, HDQ);
  pad_convert_kernel<<<cdiv((size_t)GQ*HDQ,256),256,0,stream>>>(w_hh1f, WHH1F, GQ, HDQ, GQ, HDQ);
  pad_convert_kernel<<<cdiv((size_t)GQ*HDQ,256),256,0,stream>>>(w_hh1b, WHH1B, GQ, HDQ, GQ, HDQ);
  pad_convert_kernel<<<cdiv((size_t)128*HQ,256),256,0,stream>>>(w_out, WOUT, LQ, HQ, 128, HQ);
  bias_sum_kernel<<<cdiv(GQ,256),256,0,stream>>>(b_ih0f, b_hh0f, BIAS0F, GQ);
  bias_sum_kernel<<<cdiv(GQ,256),256,0,stream>>>(b_ih0b, b_hh0b, BIAS0B, GQ);
  bias_sum_kernel<<<cdiv(GQ,256),256,0,stream>>>(b_ih1f, b_hh1f, BIAS1F, GQ);
  bias_sum_kernel<<<cdiv(GQ,256),256,0,stream>>>(b_ih1b, b_hh1b, BIAS1B, GQ);
  pad_bias_kernel<<<1,16,0,stream>>>(b_out, BOUT16, LQ, 16);

  // --- embedding gather ---
  embed_kernel<<<cdiv((size_t)MR*EP,256),256,0,stream>>>(ids, emb, X0);

  // --- layer 0: input projections, then scans ---
  dim3 gridG(MR / 64, GQ / 128);
  gemm_bf16_kernel<<<gridG,256,0,stream>>>(X0, WIH0F, BIAS0F, GF, EP, EP, EP, GQ, GQ);
  gemm_bf16_kernel<<<gridG,256,0,stream>>>(X0, WIH0B, BIAS0B, GB, EP, EP, EP, GQ, GQ);
  lstm_scan_kernel<<<1,256,0,stream>>>(GF, WHH0F, H0, HQ, 0,   0);
  lstm_scan_kernel<<<1,256,0,stream>>>(GB, WHH0B, H0, HQ, HDQ, 1);

  // --- layer 1 ---
  gemm_bf16_kernel<<<gridG,256,0,stream>>>(H0, WIH1F, BIAS1F, GF, HQ, HQ, HQ, GQ, GQ);
  gemm_bf16_kernel<<<gridG,256,0,stream>>>(H0, WIH1B, BIAS1B, GB, HQ, HQ, HQ, GQ, GQ);
  lstm_scan_kernel<<<1,256,0,stream>>>(GF, WHH1F, H1, HQ, 0,   0);
  lstm_scan_kernel<<<1,256,0,stream>>>(GB, WHH1B, H1, HQ, HDQ, 1);

  // --- emissions ---
  dim3 gridE(MR / 64, 1);
  gemm_bf16_kernel<<<gridE,256,0,stream>>>(H1, WOUT, BOUT16, EM, HQ, HQ, HQ, 16, 16);
  emcopy_kernel<<<cdiv((size_t)MR*LQ,256),256,0,stream>>>(EM, out + 1);

  // --- CRF ---
  dim3 crfBlk(BQ, LQ);
  crf_logz_kernel<<<1,crfBlk,0,stream>>>(EM, amask, c_start, c_trans, c_end, LOGZ);
  crf_score_loss_kernel<<<1,BQ,0,stream>>>(EM, labels, amask, c_start, c_trans, c_end,
                                           LOGZ, out);
  viterbi_fwd_kernel<<<1,crfBlk,0,stream>>>(EM, amask, c_start, c_trans, c_end, BP, LAST);
  viterbi_back_kernel<<<1,BQ,0,stream>>>(BP, LAST, amask, out + 1 + (size_t)MR * LQ);
}
